// MambaModel_16870631539012
// MI455X (gfx1250) — compile-verified, matrix operand order checked by reading, not run on hardware
//
#include <hip/hip_runtime.h>
#include <hip/hip_bf16.h>
#include <math.h>
#include <stdint.h>

// ---------------- model dims ----------------
#define VOCAB    64
#define D_MODEL  1024
#define D_STATE  128
#define DCONV    4
#define HEADDIM  64
#define N_LAYERS 4
#define BATCH    4
#define SEQLEN   4096
#define D_INNER  2048
#define NHEADS   32
#define CONV_DIM 2304      // D_INNER + 2*D_STATE
#define D_PROJ   4384      // 2*D_INNER + 2*D_STATE + NHEADS
#define TOK      (BATCH * SEQLEN)
#define EPSF     1e-5f

#define USE_TDM 1          // flip to 0 to fall back to manual global->LDS staging

typedef __attribute__((ext_vector_type(8)))  __bf16 v8bf;
typedef __attribute__((ext_vector_type(16))) __bf16 v16bf;
typedef __attribute__((ext_vector_type(8)))  float  v8f;
typedef __attribute__((ext_vector_type(4)))  unsigned v4u;
typedef __attribute__((ext_vector_type(8)))  unsigned v8u;

__device__ __forceinline__ unsigned short f2bf(float f) {
    unsigned u = __float_as_uint(f);
    u += 0x7FFFu + ((u >> 16) & 1u);   // round-to-nearest-even
    return (unsigned short)(u >> 16);
}
__device__ __forceinline__ float siluf(float x) { return x / (1.f + __expf(-x)); }

// 256-thread block sum reduction (wave32)
__device__ __forceinline__ float block_reduce_sum(float v) {
    #pragma unroll
    for (int o = 16; o > 0; o >>= 1) v += __shfl_down(v, o, 32);
    __shared__ float ws[8];
    __shared__ float total;
    int lane = threadIdx.x & 31, w = threadIdx.x >> 5;
    if (lane == 0) ws[w] = v;
    __syncthreads();
    if (threadIdx.x == 0) {
        float r = 0.f;
        #pragma unroll
        for (int i = 0; i < 8; ++i) r += ws[i];
        total = r;
    }
    __syncthreads();
    return total;
}

// ---------------- elementwise kernels ----------------
__global__ void cast_bf16_kernel(const float* __restrict__ in,
                                 unsigned short* __restrict__ out, int n) {
    int i = blockIdx.x * blockDim.x + threadIdx.x;
    int stride = gridDim.x * blockDim.x;
    for (; i < n; i += stride) out[i] = f2bf(in[i]);
}

__global__ void embed_kernel(const int* __restrict__ x, const float* __restrict__ emb,
                             float* __restrict__ h) {
    int tok = blockIdx.x;
    const float* src = emb + (size_t)x[tok] * D_MODEL;
    float* dst = h + (size_t)tok * D_MODEL;
    for (int i = threadIdx.x; i < D_MODEL; i += blockDim.x) dst[i] = src[i];
}

__global__ void __launch_bounds__(256)
rmsnorm_bf16_kernel(const float* __restrict__ x, const float* __restrict__ w,
                    unsigned short* __restrict__ out, int D) {
    size_t base = (size_t)blockIdx.x * D;
    float ss = 0.f;
    for (int i = threadIdx.x; i < D; i += blockDim.x) { float v = x[base + i]; ss += v * v; }
    ss = block_reduce_sum(ss);
    float r = rsqrtf(ss / (float)D + EPSF);
    for (int i = threadIdx.x; i < D; i += blockDim.x)
        out[base + i] = f2bf(x[base + i] * r * w[i]);
}

// depthwise causal conv (width 4) over the xBC slice of zxbcdt, + SiLU
__global__ void __launch_bounds__(256)
conv_silu_kernel(const float* __restrict__ zx, const float* __restrict__ cw,
                 const float* __restrict__ cb, float* __restrict__ xbc) {
    int tok = blockIdx.x;
    int b = tok / SEQLEN, l = tok % SEQLEN;
    #pragma unroll
    for (int j = 0; j < 9; ++j) {           // 9*256 = 2304 channels
        int c = threadIdx.x + j * 256;
        float acc = cb[c];
        #pragma unroll
        for (int k = 0; k < DCONV; ++k) {
            int ll = l + k - (DCONV - 1);
            if (ll >= 0)
                acc += zx[((size_t)(b * SEQLEN + ll)) * D_PROJ + D_INNER + c] * cw[c * DCONV + k];
        }
        xbc[(size_t)tok * CONV_DIM + c] = siluf(acc);
    }
}

// dt = softplus(dt_raw + bias); dA = exp(dt * -exp(A_log))
__global__ void dt_kernel(const float* __restrict__ zx, const float* __restrict__ dtb,
                          const float* __restrict__ alog,
                          float* __restrict__ dt, float* __restrict__ dA) {
    int i = blockIdx.x * blockDim.x + threadIdx.x;
    if (i >= TOK * NHEADS) return;
    int h = i % NHEADS;
    size_t tok = (size_t)(i / NHEADS);
    float v = zx[tok * D_PROJ + D_INNER + CONV_DIM + h] + dtb[h];
    float sp = (v > 20.f) ? v : log1pf(__expf(v));
    dt[i] = sp;
    dA[i] = __expf(sp * (-__expf(alog[h])));
}

// sequential selective-state scan: one block per (batch, head); state 64x128 in regs
__global__ void __launch_bounds__(256)
scan_kernel(const float* __restrict__ xbc, const float* __restrict__ dtbuf,
            const float* __restrict__ dAbuf, const float* __restrict__ Dp,
            float* __restrict__ y) {
    int b  = blockIdx.x / NHEADS;
    int hh = blockIdx.x % NHEADS;
    __shared__ float sB[D_STATE], sC[D_STATE], sx[HEADDIM], sy[256];
    float hs[32];
    #pragma unroll
    for (int i = 0; i < 32; ++i) hs[i] = 0.f;
    const int p = threadIdx.x & 63;          // head-dim index
    const int ng = threadIdx.x >> 6;         // state-group 0..3 (32 n each)
    const int nbase = ng * 32;
    const float Dh = Dp[hh];
    const int t = threadIdx.x;
    for (int l = 0; l < SEQLEN; ++l) {
        size_t tok = (size_t)b * SEQLEN + l;
        const float* row = xbc + tok * CONV_DIM;
        if (t < D_STATE) sB[t] = row[D_INNER + t];
        else             sC[t - D_STATE] = row[D_INNER + D_STATE + (t - D_STATE)];
        if (t < HEADDIM) sx[t] = row[hh * HEADDIM + t];
        __syncthreads();
        float dt = dtbuf[tok * NHEADS + hh];
        float dA = dAbuf[tok * NHEADS + hh];
        float dtx = dt * sx[p];
        float yp = 0.f;
        #pragma unroll
        for (int i = 0; i < 32; ++i) {
            float hv = dA * hs[i] + dtx * sB[nbase + i];
            hs[i] = hv;
            yp += hv * sC[nbase + i];
        }
        sy[ng * 64 + p] = yp;
        __syncthreads();
        if (t < HEADDIM) {
            float v = sy[p] + sy[64 + p] + sy[128 + p] + sy[192 + p] + Dh * sx[p];
            y[tok * D_INNER + hh * HEADDIM + p] = v;
        }
        __syncthreads();
    }
}

// y' = rmsnorm(y * silu(z), gw) -> bf16   (D_INNER = 2048, 256 threads * 8)
__global__ void __launch_bounds__(256)
gated_rmsnorm_kernel(const float* __restrict__ y, const float* __restrict__ zx,
                     const float* __restrict__ gw, unsigned short* __restrict__ out) {
    size_t tok = blockIdx.x;
    const float* yrow = y + tok * (size_t)D_INNER;
    const float* zrow = zx + tok * (size_t)D_PROJ;   // z = first D_INNER of zxbcdt
    float vals[8];
    float ss = 0.f;
    #pragma unroll
    for (int j = 0; j < 8; ++j) {
        int i = threadIdx.x + j * 256;
        float v = yrow[i] * siluf(zrow[i]);
        vals[j] = v; ss += v * v;
    }
    ss = block_reduce_sum(ss);
    float r = rsqrtf(ss / (float)D_INNER + EPSF);
    #pragma unroll
    for (int j = 0; j < 8; ++j) {
        int i = threadIdx.x + j * 256;
        out[tok * D_INNER + i] = f2bf(vals[j] * r * gw[i]);
    }
}

// ---------------- WMMA bf16 GEMM: C[M,N] = A[M,K] * W[N,K]^T (+ addend) ----------------
#define BM 256
#define BN 64
#define BK 32
#define LDS_STRIDE 40   // 32 payload + 4-DWORD TDM pad -> 80B rows, 16B-aligned

#if USE_TDM
// Issue a 2D TDM tile load: tile_rows x 32 bf16 elements, row stride = row_stride
// elements in global memory, zero-fill rows beyond avail_rows. LDS gets 64B rows
// padded by 16B (pad_interval code 3 = 64B, pad_amount code 3 = 4 DWORDs).
__device__ __forceinline__ void tdm_load_2d(unsigned lds_addr, const void* gaddr,
                                            unsigned tile_rows, unsigned avail_rows,
                                            unsigned row_stride) {
    unsigned long long ga = (unsigned long long)(uintptr_t)gaddr;
    v4u g0; v8u g1;
    g0[0] = 1u;                                   // count=1, user descriptor
    g0[1] = lds_addr;                             // D# bits 63:32
    g0[2] = (unsigned)ga;                         // global_addr[31:0]
    g0[3] = (unsigned)(ga >> 32) | (2u << 30);    // global_addr[56:32] | type=2
    const unsigned td0 = 0x40000000u;             // columns always in-bounds
    const unsigned td1 = avail_rows;              // rows beyond -> zero-fill
    g1[0] = (1u << 16)                            // data_size = 2 bytes
          | (1u << 20)                            // pad_enable
          | (3u << 22)                            // pad_interval: 16 DWORDs (64B)
          | (3u << 25);                           // pad_amount: 4 DWORDs (16B)
    g1[1] = (td0 & 0xFFFFu) << 16;                // tensor_dim0[15:0]
    g1[2] = (td0 >> 16) | ((td1 & 0xFFFFu) << 16);// tensor_dim0[31:16] | tensor_dim1[15:0]
    g1[3] = (td1 >> 16) | ((unsigned)BK << 16);   // tensor_dim1[31:16] | tile_dim0=32
    g1[4] = tile_rows;                            // tile_dim1 | tile_dim2=0
    g1[5] = row_stride;                           // tensor_dim0_stride[31:0]
    g1[6] = 0u;                                   // stride[47:32] | dim1_stride[15:0]
    g1[7] = 0u;
    asm volatile("tensor_load_to_lds %0, %1" :: "s"(g0), "s"(g1) : "memory");
}
#endif

// one K-step of WMMA work from a given LDS buffer pair (fragments per ISA 7.12.2)
__device__ __forceinline__ void gemm_compute_step(const unsigned short* bufA,
                                                  const unsigned short* bufB,
                                                  int lane, int wave, v8f acc[2][4]) {
    const int kb = (lane >> 4) * 8;
    v16bf af[2];
    #pragma unroll
    for (int mi = 0; mi < 2; ++mi) {
        const int rowA = wave * 32 + mi * 16 + (lane & 15);
        v8bf alo = *reinterpret_cast<const v8bf*>(&bufA[rowA * LDS_STRIDE + kb]);
        v8bf ahi = *reinterpret_cast<const v8bf*>(&bufA[rowA * LDS_STRIDE + kb + 16]);
        af[mi] = __builtin_shufflevector(alo, ahi,
                     0,1,2,3,4,5,6,7,8,9,10,11,12,13,14,15);
    }
    #pragma unroll
    for (int j = 0; j < 4; ++j) {
        const int rowB = j * 16 + (lane & 15);
        v8bf blo = *reinterpret_cast<const v8bf*>(&bufB[rowB * LDS_STRIDE + kb]);
        v8bf bhi = *reinterpret_cast<const v8bf*>(&bufB[rowB * LDS_STRIDE + kb + 16]);
        v16bf bfv = __builtin_shufflevector(blo, bhi,
                        0,1,2,3,4,5,6,7,8,9,10,11,12,13,14,15);
        acc[0][j] = __builtin_amdgcn_wmma_f32_16x16x32_bf16(
                        false, af[0], false, bfv, (short)0, acc[0][j], false, false);
        acc[1][j] = __builtin_amdgcn_wmma_f32_16x16x32_bf16(
                        false, af[1], false, bfv, (short)0, acc[1][j], false, false);
    }
}

__global__ void __launch_bounds__(256)
wmma_gemm_kernel(const unsigned short* __restrict__ A,   // bf16 [M,K]
                 const unsigned short* __restrict__ W,   // bf16 [N,K]
                 const float* __restrict__ addend,       // nullable [M,N]
                 float* __restrict__ C,                  // f32 [M,N]
                 int M, int N, int K) {
    // double-buffered tiles: 2 * (20480 + 5120) = 51200 B of LDS
    __shared__ __align__(16) unsigned short lA[2][BM * LDS_STRIDE];
    __shared__ __align__(16) unsigned short lB[2][BN * LDS_STRIDE];
    const int n0 = blockIdx.x * BN;
    const int m0 = blockIdx.y * BM;
    const int t = threadIdx.x;
    const int lane = t & 31;
    const int wave = t >> 5;     // 8 waves; wave owns M rows [wave*32, wave*32+32)

    v8f acc[2][4] = {};
    const int nK = K / BK;

#if USE_TDM
    const unsigned rowsB = (unsigned)(N - n0);       // grid guarantees > 0
    // prologue: kick off tile 0 with the Tensor Data Mover (wave 0 only)
    if (t < 32) {
        tdm_load_2d((unsigned)(uintptr_t)lA[0], A + (size_t)m0 * K, BM, 0x40000000u, (unsigned)K);
        tdm_load_2d((unsigned)(uintptr_t)lB[0], W + (size_t)n0 * K, BN, rowsB,       (unsigned)K);
    }
    int cur = 0;
    for (int kt = 0; kt < nK; ++kt) {
        if (t < 32) __builtin_amdgcn_s_wait_tensorcnt(0);   // tile kt landed
        __syncthreads();   // tile kt visible; prior reads of the other buffer done
        if (t < 32 && (kt + 1) < nK) {   // overlap: fetch tile kt+1 while computing kt
            size_t k0 = (size_t)(kt + 1) * BK;
            tdm_load_2d((unsigned)(uintptr_t)lA[cur ^ 1], A + (size_t)m0 * K + k0,
                        BM, 0x40000000u, (unsigned)K);
            tdm_load_2d((unsigned)(uintptr_t)lB[cur ^ 1], W + (size_t)n0 * K + k0,
                        BN, rowsB, (unsigned)K);
        }
        gemm_compute_step(lA[cur], lB[cur], lane, wave, acc);
        cur ^= 1;
    }
#else
    for (int kt = 0; kt < nK; ++kt) {
        int k0 = kt * BK;
        #pragma unroll
        for (int i = 0; i < 4; ++i) {                 // A: 256x32 bf16 = 4 x uint4/thread
            int idx = t + i * 256;
            int r = idx >> 2;
            int cc = (idx & 3) * 8;
            const uint4 v = *reinterpret_cast<const uint4*>(A + (size_t)(m0 + r) * K + k0 + cc);
            *reinterpret_cast<uint4*>(&lA[0][r * LDS_STRIDE + cc]) = v;
        }
        {
            int r = t >> 2;
            int cc = (t & 3) * 8;
            int n = n0 + r;
            uint4 v = make_uint4(0u, 0u, 0u, 0u);
            if (n < N) v = *reinterpret_cast<const uint4*>(W + (size_t)n * K + k0 + cc);
            *reinterpret_cast<uint4*>(&lB[0][r * LDS_STRIDE + cc]) = v;
        }
        __syncthreads();
        gemm_compute_step(lA[0], lB[0], lane, wave, acc);
        __syncthreads();
    }
#endif

    // store: VGPR r -> M = r + 8*(lane>=16); N = lane&15 within each 16x16 tile
    #pragma unroll
    for (int mi = 0; mi < 2; ++mi) {
        const int mbase = m0 + wave * 32 + mi * 16 + (lane >> 4) * 8;
        #pragma unroll
        for (int j = 0; j < 4; ++j) {
            int gn = n0 + j * 16 + (lane & 15);
            if (gn < N) {
                #pragma unroll
                for (int r = 0; r < 8; ++r) {
                    size_t idx = (size_t)(mbase + r) * N + gn;
                    float v = acc[mi][j][r];
                    if (addend) v += addend[idx];
                    C[idx] = v;
                }
            }
        }
    }
}

// ---------------- host orchestration ----------------
extern "C" void kernel_launch(void* const* d_in, const int* in_sizes, int n_in,
                              void* d_out, int out_size, void* d_ws, size_t ws_size,
                              hipStream_t stream) {
    (void)in_sizes; (void)n_in; (void)out_size; (void)ws_size;
    const int*   x     = (const int*)d_in[0];
    const float* emb   = (const float*)d_in[1];
    const float* w_in  = (const float*)d_in[2];
    const float* cw    = (const float*)d_in[3];
    const float* cb    = (const float*)d_in[4];
    const float* dtb   = (const float*)d_in[5];
    const float* alog  = (const float*)d_in[6];
    const float* Dp    = (const float*)d_in[7];
    const float* gw    = (const float*)d_in[8];
    const float* w_out = (const float*)d_in[9];
    const float* rmsw  = (const float*)d_in[10];
    const float* normf = (const float*)d_in[11];
    const float* w_lm  = (const float*)d_in[12];

    char* ws = (char*)d_ws;
    size_t off = 0;
    auto alloc = [&](size_t bytes) -> char* {
        char* p = ws + off;
        off += (bytes + 255) & ~(size_t)255;
        return p;
    };
    float* hA            = (float*)alloc((size_t)TOK * D_MODEL * 4);
    float* hB            = (float*)alloc((size_t)TOK * D_MODEL * 4);
    unsigned short* xn   = (unsigned short*)alloc((size_t)TOK * D_MODEL * 2);
    float* zx            = (float*)alloc((size_t)TOK * D_PROJ * 4);
    float* xbc           = (float*)alloc((size_t)TOK * CONV_DIM * 4);
    float* dt            = (float*)alloc((size_t)TOK * NHEADS * 4);
    float* dA            = (float*)alloc((size_t)TOK * NHEADS * 4);
    float* ybuf          = (float*)alloc((size_t)TOK * D_INNER * 4);
    unsigned short* ybf  = (unsigned short*)alloc((size_t)TOK * D_INNER * 2);
    unsigned short* wibf = (unsigned short*)alloc((size_t)N_LAYERS * D_PROJ * D_MODEL * 2);
    unsigned short* wobf = (unsigned short*)alloc((size_t)N_LAYERS * D_MODEL * D_INNER * 2);
    unsigned short* wlbf = (unsigned short*)alloc((size_t)VOCAB * D_MODEL * 2);

    // weights -> bf16 (recomputed every call; deterministic)
    cast_bf16_kernel<<<1024, 256, 0, stream>>>(w_in,  wibf, N_LAYERS * D_PROJ * D_MODEL);
    cast_bf16_kernel<<<1024, 256, 0, stream>>>(w_out, wobf, N_LAYERS * D_MODEL * D_INNER);
    cast_bf16_kernel<<<64,   256, 0, stream>>>(w_lm,  wlbf, VOCAB * D_MODEL);

    embed_kernel<<<TOK, 256, 0, stream>>>(x, emb, hA);

    float* hcur = hA; float* hnext = hB;
    for (int i = 0; i < N_LAYERS; ++i) {
        rmsnorm_bf16_kernel<<<TOK, 256, 0, stream>>>(hcur, rmsw + i * D_MODEL, xn, D_MODEL);

        dim3 g1((D_PROJ + BN - 1) / BN, TOK / BM);
        wmma_gemm_kernel<<<g1, 256, 0, stream>>>(xn, wibf + (size_t)i * D_PROJ * D_MODEL,
                                                 nullptr, zx, TOK, D_PROJ, D_MODEL);

        conv_silu_kernel<<<TOK, 256, 0, stream>>>(zx, cw + (size_t)i * CONV_DIM * DCONV,
                                                  cb + (size_t)i * CONV_DIM, xbc);
        dt_kernel<<<(TOK * NHEADS + 255) / 256, 256, 0, stream>>>(
            zx, dtb + i * NHEADS, alog + i * NHEADS, dt, dA);

        scan_kernel<<<BATCH * NHEADS, 256, 0, stream>>>(xbc, dt, dA, Dp + i * NHEADS, ybuf);

        gated_rmsnorm_kernel<<<TOK, 256, 0, stream>>>(ybuf, zx, gw + (size_t)i * D_INNER, ybf);

        dim3 g2((D_MODEL + BN - 1) / BN, TOK / BM);
        wmma_gemm_kernel<<<g2, 256, 0, stream>>>(ybf, wobf + (size_t)i * D_MODEL * D_INNER,
                                                 hcur, hnext, TOK, D_MODEL, D_INNER);
        float* tmp = hcur; hcur = hnext; hnext = tmp;
    }

    rmsnorm_bf16_kernel<<<TOK, 256, 0, stream>>>(hcur, normf, xn, D_MODEL);
    dim3 g3((VOCAB + BN - 1) / BN, TOK / BM);
    wmma_gemm_kernel<<<g3, 256, 0, stream>>>(xn, wlbf, nullptr, (float*)d_out,
                                             TOK, VOCAB, D_MODEL);
}
